// EnhancedJointer_31396210934102
// MI455X (gfx1250) — matrix operand order repaired and checked by reading, not compiled, not run on hardware
//
#include <hip/hip_runtime.h>
#include <hip/hip_bf16.h>

// ---------------- problem dims ----------------
#define BB 8
#define TT 128
#define UU 64
#define EE 768
#define HH 512
#define VV 1024
#define EPS 1e-5f

typedef __attribute__((ext_vector_type(16))) __bf16 v16bf;
typedef __attribute__((ext_vector_type(8)))  float  v8f;

union AFrag { v16bf v; uint4 q[2]; };

// LDS strides (elements), padded to dodge bank conflicts
#define A_STRIDE 784   // bf16: holds up to 64x768 (comb / final)
#define H_STRIDE 528   // bf16: holds 64x512 (f / h)
#define C_STRIDE 516   // f32 : holds 64x512 GEMM accum scratch

// ---------------- WMMA fragment loads ----------------
// A-matrix 16x32 bf16 (ISA 7.12.2): lane l holds row m0+(l&15);
// elems 0..7  <-> K = k0 + 8*(l>>4) + {0..7}
// elems 8..15 <-> K = k0 + 16 + 8*(l>>4) + {0..7}
__device__ __forceinline__ void load_afrag(const __bf16* A, int stride, int m0, int k0,
                                           int lane, AFrag& f) {
    int m = m0 + (lane & 15);
    int half = lane >> 4;
    const __bf16* base = A + m * stride + k0 + half * 8;
    f.q[0] = *(const uint4*)(base);
    f.q[1] = *(const uint4*)(base + 16);
}

// B-matrix 32x16 bf16: lane l holds col n0+(l&15);
// elems 0..7 <-> K = k0 + 8*(l>>4) + e; elems 8..15 <-> K = k0 + 16 + 8*(l>>4) + (e-8)
__device__ __forceinline__ void load_bfrag(const __bf16* W, int K, int n0, int k0,
                                           int lane, AFrag& f) {
    int n = n0 + (lane & 15);
    int half = lane >> 4;
    const __bf16* base = W + (size_t)n * K + k0 + half * 8;
    f.q[0] = *(const uint4*)(base);
    f.q[1] = *(const uint4*)(base + 16);
}

__device__ __forceinline__ v8f wmma_bf16(const AFrag& a, const AFrag& b, v8f c) {
    return __builtin_amdgcn_wmma_f32_16x16x32_bf16(false, a.v, false, b.v,
                                                   (short)0, c, false, false);
}

// GEMM: C[64 x NTILES*16] += A[64 x K] * W^T slab, per-wave n0
template <int NTILES>
__device__ __forceinline__ void gemm_tiles(const __bf16* A, int aStride,
                                           const __bf16* W, int K, int n0,
                                           int lane, v8f acc[4][NTILES]) {
    for (int kk = 0; kk < K; kk += 32) {
        AFrag bf[NTILES];
        #pragma unroll
        for (int j = 0; j < NTILES; ++j) load_bfrag(W, K, n0 + 16 * j, kk, lane, bf[j]);
        #pragma unroll
        for (int mt = 0; mt < 4; ++mt) {
            AFrag af;
            load_afrag(A, aStride, mt * 16, kk, lane, af);
            #pragma unroll
            for (int j = 0; j < NTILES; ++j) acc[mt][j] = wmma_bf16(af, bf[j], acc[mt][j]);
        }
    }
}

// Store C tiles (+bias) into f32 LDS scratch. C/D layout: M = i + 8*(lane>>4), N = lane&15.
template <int NTILES>
__device__ __forceinline__ void store_c(float* sc, const float* bias, int n0,
                                        int lane, v8f acc[4][NTILES]) {
    int half = lane >> 4, nl = lane & 15;
    #pragma unroll
    for (int j = 0; j < NTILES; ++j) {
        int col = n0 + 16 * j + nl;
        float bb = bias[col];
        #pragma unroll
        for (int mt = 0; mt < 4; ++mt)
            #pragma unroll
            for (int i = 0; i < 8; ++i)
                sc[(mt * 16 + i + 8 * half) * C_STRIDE + col] = acc[mt][j][i] + bb;
    }
}

// LayerNorm + ReLU over N cols of 64 rows in f32 scratch -> bf16 dest. 16 waves x 4 rows.
__device__ __forceinline__ void ln_relu(const float* sc, int N, const float* g,
                                        const float* bn, __bf16* dst, int dstStride,
                                        int wid, int lane) {
    float invN = 1.0f / (float)N;
    for (int r = 0; r < 4; ++r) {
        int row = wid * 4 + r;
        float s = 0.f, q = 0.f;
        for (int c = lane; c < N; c += 32) {
            float x = sc[row * C_STRIDE + c];
            s += x; q += x * x;
        }
        #pragma unroll
        for (int off = 16; off; off >>= 1) {
            s += __shfl_xor(s, off, 32);
            q += __shfl_xor(q, off, 32);
        }
        float mean = s * invN;
        float var  = q * invN - mean * mean;
        float inv  = rsqrtf(var + EPS);
        for (int c = lane; c < N; c += 32) {
            float x = sc[row * C_STRIDE + c];
            float y = (x - mean) * inv * g[c] + bn[c];
            dst[row * dstStride + c] = (__bf16)(y > 0.f ? y : 0.f);
        }
    }
}

// ---------------- small prolog kernels ----------------
__global__ void f32_to_bf16_kernel(const float* __restrict__ src,
                                   __bf16* __restrict__ dst, int n) {
    int i = blockIdx.x * blockDim.x + threadIdx.x;
    if (i < n) dst[i] = (__bf16)src[i];
}

// y = relu(LN(x @ W^T + b)); one block per row, K in, 512 out
__global__ __launch_bounds__(256) void proj_ln_relu_kernel(
    const float* __restrict__ X, int K, const float* __restrict__ W,
    const float* __restrict__ bias, const float* __restrict__ g,
    const float* __restrict__ bn, float* __restrict__ Y) {
    __shared__ float sx[EE];
    __shared__ float sy[HH];
    __shared__ float red[8], red2[8];
    int row = blockIdx.x, tid = threadIdx.x;
    const float* x = X + (size_t)row * K;
    for (int i = tid; i < K; i += 256) sx[i] = x[i];
    __syncthreads();
    for (int o = tid; o < HH; o += 256) {
        const float* w = W + (size_t)o * K;
        float acc = bias[o];
        for (int k = 0; k < K; k += 4) {
            float4 wv = *(const float4*)(w + k);
            acc += sx[k] * wv.x + sx[k + 1] * wv.y + sx[k + 2] * wv.z + sx[k + 3] * wv.w;
        }
        sy[o] = acc;
    }
    __syncthreads();
    float s = 0.f, q = 0.f;
    for (int o = tid; o < HH; o += 256) { float v = sy[o]; s += v; q += v * v; }
    #pragma unroll
    for (int off = 16; off; off >>= 1) { s += __shfl_xor(s, off, 32); q += __shfl_xor(q, off, 32); }
    int wid = tid >> 5, lane = tid & 31;
    if (lane == 0) { red[wid] = s; red2[wid] = q; }
    __syncthreads();
    if (tid == 0) {
        float ts = 0.f, tq = 0.f;
        for (int i = 0; i < 8; ++i) { ts += red[i]; tq += red2[i]; }
        red[0] = ts; red2[0] = tq;
    }
    __syncthreads();
    float mean = red[0] / (float)HH;
    float var  = red2[0] / (float)HH - mean * mean;
    float inv  = rsqrtf(var + EPS);
    for (int o = tid; o < HH; o += 256) {
        float v = (sy[o] - mean) * inv * g[o] + bn[o];
        Y[(size_t)row * HH + o] = v > 0.f ? v : 0.f;
    }
}

// att_u = (dec_p @ Wv^T + bv) @ Wo^T + bo ; one block per (b,u) row; bf16 out
__global__ __launch_bounds__(256) void att_proj_kernel(
    const float* __restrict__ decP, const float* __restrict__ Wv,
    const float* __restrict__ bv, const float* __restrict__ Wo,
    const float* __restrict__ bo, __bf16* __restrict__ attb) {
    __shared__ float srow[HH];
    __shared__ float st1[HH];
    int row = blockIdx.x, tid = threadIdx.x;
    for (int i = tid; i < HH; i += 256) srow[i] = decP[(size_t)row * HH + i];
    __syncthreads();
    for (int o = tid; o < HH; o += 256) {
        const float* w = Wv + (size_t)o * HH;
        float a = bv[o];
        for (int k = 0; k < HH; k += 4) {
            float4 wv = *(const float4*)(w + k);
            a += srow[k] * wv.x + srow[k + 1] * wv.y + srow[k + 2] * wv.z + srow[k + 3] * wv.w;
        }
        st1[o] = a;
    }
    __syncthreads();
    for (int o = tid; o < HH; o += 256) {
        const float* w = Wo + (size_t)o * HH;
        float a = bo[o];
        for (int k = 0; k < HH; k += 4) {
            float4 wv = *(const float4*)(w + k);
            a += st1[k] * wv.x + st1[k + 1] * wv.y + st1[k + 2] * wv.z + st1[k + 3] * wv.w;
        }
        attb[(size_t)row * HH + o] = (__bf16)a;
    }
}

// ---------------- fused mega kernel: one WG per (b,t), M = 64 rows (all u) ----------------
__global__ __launch_bounds__(512) void jointer_mega_kernel(
    const float* __restrict__ encP, const float* __restrict__ decP,
    const __bf16* __restrict__ attb,
    const __bf16* __restrict__ Wf1b, const float* __restrict__ bf1,
    const float* __restrict__ gf1, const float* __restrict__ bnf1,
    const __bf16* __restrict__ Wf2b, const float* __restrict__ bf2,
    const float* __restrict__ gf2, const float* __restrict__ bnf2,
    const __bf16* __restrict__ W1b, const float* __restrict__ b1,
    const float* __restrict__ g1, const float* __restrict__ bn1,
    const __bf16* __restrict__ W2b, const float* __restrict__ b2,
    const float* __restrict__ ssw, float* __restrict__ out) {
    extern __shared__ char smem[];
    __bf16* s_a = (__bf16*)smem;                                   // 64 x A_STRIDE bf16
    float*  s_c = (float*)(smem + 64 * A_STRIDE * 2);              // 64 x C_STRIDE f32
    __bf16* s_h = (__bf16*)(smem + 64 * A_STRIDE * 2 + 64 * C_STRIDE * 4); // 64 x H_STRIDE bf16

    const int tid = threadIdx.x;
    const int lane = tid & 31;
    const int wid = tid >> 5;
    const int bt = blockIdx.x;      // b*T + t
    const int b = bt / TT;

    const float* encRow = encP + (size_t)bt * HH;
    const float* decB   = decP + (size_t)b * UU * HH;

    // ---- comb = enc_p[b,t] + dec_p[b,u]  -> bf16 A ----
    for (int j = tid; j < UU * HH; j += 512) {
        int u = j >> 9, k = j & (HH - 1);
        s_a[u * A_STRIDE + k] = (__bf16)(encRow[k] + decB[u * HH + k]);
    }
    __syncthreads();

    // ---- GEMM1: f_raw = comb @ Wf1^T + bf1   (N=512, K=512) ----
    {
        int n0 = wid * 32;
        v8f acc[4][2] = {};
        gemm_tiles<2>(s_a, A_STRIDE, Wf1b, HH, n0, lane, acc);
        store_c<2>(s_c, bf1, n0, lane, acc);
    }
    __syncthreads();
    ln_relu(s_c, HH, gf1, bnf1, s_h, H_STRIDE, wid, lane);   // f -> s_h (bf16)
    __syncthreads();

    // ---- GEMM2: fused_raw = f @ Wf2^T + bf2  (N=256, K=512) ----
    {
        int n0 = wid * 16;
        v8f acc[4][1] = {};
        gemm_tiles<1>(s_h, H_STRIDE, Wf2b, HH, n0, lane, acc);
        store_c<1>(s_c, bf2, n0, lane, acc);
    }
    __syncthreads();
    ln_relu(s_c, HH / 2, gf2, bnf2, s_a, A_STRIDE, wid, lane); // fused -> s_a[:, 0:256]
    // concat att into s_a[:, 256:768]
    for (int j = tid; j < UU * HH; j += 512) {
        int u = j >> 9, c = j & (HH - 1);
        s_a[u * A_STRIDE + (HH / 2) + c] = attb[((size_t)(b * UU + u)) * HH + c];
    }
    __syncthreads();

    // ---- GEMM3: h_raw = final @ W1^T + b1  (N=512, K=768) ----
    {
        int n0 = wid * 32;
        v8f acc[4][2] = {};
        gemm_tiles<2>(s_a, A_STRIDE, W1b, HH / 2 + HH, n0, lane, acc);
        store_c<2>(s_c, b1, n0, lane, acc);
    }
    __syncthreads();
    ln_relu(s_c, HH, g1, bn1, s_h, H_STRIDE, wid, lane);     // h -> s_h (bf16)
    __syncthreads();

    // ---- GEMM4: logits = (h @ W2^T + b2) * ssw  (N=1024, K=512), streamed to global ----
    float* orow = out + (size_t)bt * UU * VV;
    int half = lane >> 4, nl = lane & 15;
    #pragma unroll
    for (int pass = 0; pass < 2; ++pass) {
        int n0 = wid * 64 + pass * 32;
        v8f acc[4][2] = {};
        gemm_tiles<2>(s_h, H_STRIDE, W2b, HH, n0, lane, acc);
        #pragma unroll
        for (int j = 0; j < 2; ++j) {
            int col = n0 + 16 * j + nl;
            float bb = b2[col], sc = ssw[col];
            #pragma unroll
            for (int mt = 0; mt < 4; ++mt)
                #pragma unroll
                for (int i = 0; i < 8; ++i) {
                    int row = mt * 16 + i + 8 * half;
                    orow[(size_t)row * VV + col] = (acc[mt][j][i] + bb) * sc;
                }
        }
    }
}

// ---------------- launch ----------------
extern "C" void kernel_launch(void* const* d_in, const int* in_sizes, int n_in,
                              void* d_out, int out_size, void* d_ws, size_t ws_size,
                              hipStream_t stream) {
    const float* enc = (const float*)d_in[0];
    const float* dec = (const float*)d_in[1];
    const float* We  = (const float*)d_in[2];
    const float* be  = (const float*)d_in[3];
    const float* ge  = (const float*)d_in[4];
    const float* bne = (const float*)d_in[5];
    const float* Wd  = (const float*)d_in[6];
    const float* bd  = (const float*)d_in[7];
    const float* gd  = (const float*)d_in[8];
    const float* bnd = (const float*)d_in[9];
    const float* Wf1 = (const float*)d_in[10];
    const float* bf1 = (const float*)d_in[11];
    const float* gf1 = (const float*)d_in[12];
    const float* bnf1= (const float*)d_in[13];
    const float* Wf2 = (const float*)d_in[14];
    const float* bf2 = (const float*)d_in[15];
    const float* gf2 = (const float*)d_in[16];
    const float* bnf2= (const float*)d_in[17];
    const float* Wv  = (const float*)d_in[18];
    const float* bv  = (const float*)d_in[19];
    const float* Wo  = (const float*)d_in[20];
    const float* bo  = (const float*)d_in[21];
    const float* W1  = (const float*)d_in[22];
    const float* b1  = (const float*)d_in[23];
    const float* g1  = (const float*)d_in[24];
    const float* bn1 = (const float*)d_in[25];
    const float* W2  = (const float*)d_in[26];
    const float* b2  = (const float*)d_in[27];
    const float* ssw = (const float*)d_in[28];
    float* out = (float*)d_out;

    // workspace layout (256B aligned slabs)
    char* ws = (char*)d_ws;
    size_t off = 0;
    auto alloc = [&](size_t bytes) { char* p = ws + off; off = (off + bytes + 255) & ~(size_t)255; return p; };
    float*  enc_p = (float*)alloc((size_t)BB * TT * HH * 4);   // 2 MB
    float*  dec_p = (float*)alloc((size_t)BB * UU * HH * 4);   // 1 MB
    __bf16* attb  = (__bf16*)alloc((size_t)BB * UU * HH * 2);  // 0.5 MB
    __bf16* Wf1b  = (__bf16*)alloc((size_t)HH * HH * 2);
    __bf16* Wf2b  = (__bf16*)alloc((size_t)(HH / 2) * HH * 2);
    __bf16* W1b   = (__bf16*)alloc((size_t)HH * (HH / 2 + HH) * 2);
    __bf16* W2b   = (__bf16*)alloc((size_t)VV * HH * 2);

    // weight conversions f32 -> bf16
    auto conv = [&](const float* s, __bf16* d, int n) {
        f32_to_bf16_kernel<<<(n + 255) / 256, 256, 0, stream>>>(s, d, n);
    };
    conv(Wf1, Wf1b, HH * HH);
    conv(Wf2, Wf2b, (HH / 2) * HH);
    conv(W1,  W1b,  HH * (HH / 2 + HH));
    conv(W2,  W2b,  VV * HH);

    // projections
    proj_ln_relu_kernel<<<BB * TT, 256, 0, stream>>>(enc, EE, We, be, ge, bne, enc_p);
    proj_ln_relu_kernel<<<BB * UU, 256, 0, stream>>>(dec, EE, Wd, bd, gd, bnd, dec_p);
    att_proj_kernel<<<BB * UU, 256, 0, stream>>>(dec_p, Wv, bv, Wo, bo, attb);

    // fused WMMA chain, one workgroup per (b,t)
    const size_t ldsBytes = (size_t)64 * A_STRIDE * 2 + (size_t)64 * C_STRIDE * 4 +
                            (size_t)64 * H_STRIDE * 2; // 300,032 B < 320 KB WGP LDS
    jointer_mega_kernel<<<BB * TT, 512, ldsBytes, stream>>>(
        enc_p, dec_p, attb,
        Wf1b, bf1, gf1, bnf1,
        Wf2b, bf2, gf2, bnf2,
        W1b, b1, g1, bn1,
        W2b, b2, ssw, out);
}
